// GatedLinearAttention_74105365725880
// MI455X (gfx1250) — compile-verified, matrix operand order checked by reading, not run on hardware
//
#include <hip/hip_runtime.h>
#include <hip/hip_bf16.h>

// ---------------------------------------------------------------------------
// GLA forward for MI455X (gfx1250): bf16 WMMA everywhere a matmul appears.
// D=1024, NH=4, DK=1024, DV=2048, HDK=256, HDV=512, CHUNK=56 (padded to 64),
// B=16, L=784 (28x28), NCHUNK=14.
// ---------------------------------------------------------------------------

typedef __bf16 bf16x8  __attribute__((ext_vector_type(8)));
typedef __bf16 bf16x16 __attribute__((ext_vector_type(16)));
typedef float  floatx8 __attribute__((ext_vector_type(8)));

// WMMA D = A(16x32) * B(32x16) + C, bf16 in / f32 accumulate.
static __device__ __forceinline__ floatx8 wmma_bf16(bf16x16 a, bf16x16 b, floatx8 c) {
  return __builtin_amdgcn_wmma_f32_16x16x32_bf16(false, a, false, b, (short)0, c,
                                                 false, false);
}

// A/B fragment load from LDS. Row-major [row][k] with given k-stride.
// Per the CDNA5 ISA 16-bit operand layout: lane holds row (lane&15), and its
// 16 bf16 elements map to k = {k0+8h .. k0+8h+7} and {k0+16+8h .. k0+16+8h+7}
// where h = lane>>4  ->  two contiguous 16-byte LDS reads.
static __device__ __forceinline__ bf16x16 ldsFragRM(const __bf16* base, int row, int k0,
                                                    int stride, int lane) {
  const __bf16* p = base + (size_t)(row + (lane & 15)) * stride + k0 + ((lane >> 4) << 3);
  bf16x8 lo = *(const bf16x8*)p;
  bf16x8 hi = *(const bf16x8*)(p + 16);
  bf16x16 r;
#pragma unroll
  for (int i = 0; i < 8; ++i) { r[i] = lo[i]; r[i + 8] = hi[i]; }
  return r;
}

// Same fragment but sourced from an f32 LDS tensor (running state S),
// converted to bf16 on the fly.
static __device__ __forceinline__ bf16x16 ldsFragF32(const float* base, int row, int k0,
                                                     int stride, int lane) {
  const float* p = base + (size_t)(row + (lane & 15)) * stride + k0 + ((lane >> 4) << 3);
  bf16x16 r;
#pragma unroll
  for (int i = 0; i < 8; ++i) { r[i] = (__bf16)p[i]; r[i + 8] = (__bf16)p[16 + i]; }
  return r;
}

// ---------------------------------------------------------------------------
// fp32 -> bf16 weight cast (n must be a multiple of 1024; launched that way)
// ---------------------------------------------------------------------------
__global__ void __launch_bounds__(256) cast_bf16_kernel(const float* __restrict__ s,
                                                        __bf16* __restrict__ d) {
  size_t i = ((size_t)blockIdx.x * 256 + threadIdx.x) * 4;
  float4 v = *(const float4*)(s + i);
  d[i + 0] = (__bf16)v.x;
  d[i + 1] = (__bf16)v.y;
  d[i + 2] = (__bf16)v.z;
  d[i + 3] = (__bf16)v.w;
}

// ---------------------------------------------------------------------------
// Depthwise 3x3 conv (SAME, groups=D) + SiLU.  Writes fp32 and bf16 copies.
// grid = B*28 (one block per (b,row)), 256 threads, 4 channels each.
// ---------------------------------------------------------------------------
__global__ void __launch_bounds__(256) conv_silu_kernel(const float* __restrict__ x,
                                                        const float* __restrict__ cw,
                                                        float* __restrict__ xc,
                                                        __bf16* __restrict__ xcb) {
  int by = blockIdx.x;
  int b = by / 28, y = by % 28;
  int tid = threadIdx.x;
  for (int c = tid; c < 1024; c += 256) {
    float w[9];
#pragma unroll
    for (int i = 0; i < 9; ++i) w[i] = cw[i * 1024 + c];  // (3,3,1,D)
    for (int xw = 0; xw < 28; ++xw) {
      float s = 0.f;
#pragma unroll
      for (int dy = 0; dy < 3; ++dy) {
        int yy = y + dy - 1;
        if (yy < 0 || yy > 27) continue;
#pragma unroll
        for (int dx = 0; dx < 3; ++dx) {
          int xx = xw + dx - 1;
          if (xx < 0 || xx > 27) continue;
          s += x[(size_t)(b * 784 + yy * 28 + xx) * 1024 + c] * w[dy * 3 + dx];
        }
      }
      float v = s / (1.f + __expf(-s));  // SiLU
      size_t o = (size_t)(b * 784 + y * 28 + xw) * 1024 + c;
      xc[o] = v;
      xcb[o] = (__bf16)v;
    }
  }
}

// ---------------------------------------------------------------------------
// Generic bf16 WMMA GEMM: C(f32)[M,N] = act( A(bf16)[M,K] @ B(bf16)[K,N] + bias )
// Block tile 128x64, 8 waves, each wave a 32x32 region (2x2 WMMA tiles).
// K-step 64, LDS double-buffered with register-staged prefetch: one barrier
// per stage, global loads overlap v_wmma issue.
// All global traffic is 128-bit coalesced; the B transpose to N-major happens
// on the LDS-write side (b16 scatter) since LDS bandwidth >> HBM transactions.
// grid.x = (M/128)*(N/64).  M%128==0, N%64==0, K%64==0 (true for all calls).
// ---------------------------------------------------------------------------
__global__ void __launch_bounds__(256) gemm_bf16_kernel(const __bf16* __restrict__ A,
                                                        const __bf16* __restrict__ Bw,
                                                        const float* __restrict__ bias,
                                                        float* __restrict__ C, int M,
                                                        int K, int N, int act) {
  __shared__ __bf16 As[2][128 * 64];  // row-major [m][k]
  __shared__ __bf16 Bs[2][64 * 64];   // N-major   [n][k]
  int mtiles = M >> 7;
  int bm = blockIdx.x % mtiles;
  int bn = blockIdx.x / mtiles;
  int m0 = bm * 128, n0 = bn * 64;
  int tid = threadIdx.x;
  int wave = tid >> 5, lane = tid & 31;
  int wm = (wave & 3) * 32;   // row strip within tile
  int wn = (wave >> 2) * 32;  // col strip within tile
  floatx8 acc[2][2] = {};

  bf16x8 ar[4];  // A stage: 128x64 = 8192 elems, 4 x 16B per thread
  bf16x8 br[2];  // B stage:  64x64 = 4096 elems, 2 x 16B per thread

  auto loadRegs = [&](int k0) {
#pragma unroll
    for (int it = 0; it < 4; ++it) {
      int c = tid + it * 256;              // chunk: row = c>>3, off = (c&7)*8
      int row = c >> 3, off = (c & 7) << 3;
      const __bf16* gp = A + (size_t)(m0 + row) * K + k0 + off;
      ar[it] = *(const bf16x8*)gp;
      __builtin_prefetch(gp + 64, 0, 1);   // speculative: global_prefetch_b8
    }
#pragma unroll
    for (int it = 0; it < 2; ++it) {
      int c = tid + it * 256;              // chunk: k = c>>3, nb = (c&7)*8
      int k = c >> 3, nb = (c & 7) << 3;
      const __bf16* gp = Bw + (size_t)(k0 + k) * N + n0 + nb;
      br[it] = *(const bf16x8*)gp;         // contiguous global_load_b128
    }
  };
  auto storeRegs = [&](int buf) {
#pragma unroll
    for (int it = 0; it < 4; ++it) {
      int c = tid + it * 256;
      int row = c >> 3, off = (c & 7) << 3;
      *(bf16x8*)(&As[buf][row * 64 + off]) = ar[it];
    }
#pragma unroll
    for (int it = 0; it < 2; ++it) {
      int c = tid + it * 256;
      int k = c >> 3, nb = (c & 7) << 3;
#pragma unroll
      for (int j = 0; j < 8; ++j) Bs[buf][(nb + j) * 64 + k] = br[it][j];  // transpose
    }
  };

  loadRegs(0);
  storeRegs(0);
  __syncthreads();
  int buf = 0;
  for (int k0 = 0; k0 < K; k0 += 64) {
    bool more = (k0 + 64) < K;
    if (more) loadRegs(k0 + 64);  // global loads in flight during WMMA
#pragma unroll
    for (int ks = 0; ks < 64; ks += 32)
#pragma unroll
      for (int i = 0; i < 2; ++i)
#pragma unroll
        for (int j = 0; j < 2; ++j)
          acc[i][j] = wmma_bf16(ldsFragRM(&As[buf][0], wm + i * 16, ks, 64, lane),
                                ldsFragRM(&Bs[buf][0], wn + j * 16, ks, 64, lane),
                                acc[i][j]);
    if (more) storeRegs(buf ^ 1);  // safe: prior reads of buf^1 fenced last stage
    __syncthreads();
    buf ^= 1;
  }

  int half = lane >> 4, nl = lane & 15;
#pragma unroll
  for (int i = 0; i < 2; ++i)
#pragma unroll
    for (int j = 0; j < 2; ++j) {
      int n = n0 + wn + j * 16 + nl;
      float bv = bias ? bias[n] : 0.f;
#pragma unroll
      for (int r = 0; r < 8; ++r) {
        int m = m0 + wm + i * 16 + r + half * 8;
        float v = acc[i][j][r] + bv;
        if (act == 1) v = v / (1.f + __expf(-v));  // SiLU
        C[(size_t)m * N + n] = v;
      }
    }
}

// ---------------------------------------------------------------------------
// Low-rank gate logits: gk = log_sigmoid(xc @ gk_w1 @ gk_w2 + b2) / 16
// One block per row (b,l).  Stage 1: 1024->16 via block reduction.
// Stage 2: 16->2048, split into gf / gb halves.
// ---------------------------------------------------------------------------
__global__ void __launch_bounds__(256) gk_kernel(const float* __restrict__ xc,
                                                 const float* __restrict__ w1,
                                                 const float* __restrict__ w2,
                                                 const float* __restrict__ b2,
                                                 float* __restrict__ gf,
                                                 float* __restrict__ gb) {
  __shared__ float red[256][16];
  __shared__ float y[16];
  int row = blockIdx.x;
  int tid = threadIdx.x;
  const float* xr = xc + (size_t)row * 1024;
  float p[16];
#pragma unroll
  for (int j = 0; j < 16; ++j) p[j] = 0.f;
#pragma unroll
  for (int i = 0; i < 4; ++i) {
    float xv = xr[tid * 4 + i];
#pragma unroll
    for (int j = 0; j < 16; ++j) p[j] += xv * w1[(tid * 4 + i) * 16 + j];
  }
#pragma unroll
  for (int j = 0; j < 16; ++j) red[tid][j] = p[j];
  __syncthreads();
  for (int s = 128; s > 0; s >>= 1) {
    if (tid < s) {
#pragma unroll
      for (int j = 0; j < 16; ++j) red[tid][j] += red[tid + s][j];
    }
    __syncthreads();
  }
  if (tid < 16) y[tid] = red[0][tid];
  __syncthreads();
  for (int o = tid; o < 2048; o += 256) {
    float a = b2[o];
#pragma unroll
    for (int j = 0; j < 16; ++j) a += y[j] * w2[j * 2048 + o];
    float ls = fminf(a, 0.f) - log1pf(__expf(-fabsf(a)));  // log_sigmoid
    float g = ls * (1.f / 16.f);                            // / GLN
    if (o < 1024) gf[(size_t)row * 1024 + o] = g;
    else          gb[(size_t)row * 1024 + (o - 1024)] = g;
  }
}

// ---------------------------------------------------------------------------
// Fused chunked GLA scan.  One block per (b, h, dv-tile of 128, dir).
// Running state S[dv=128][dk=256] f32 lives in LDS (128 KB) -- CDNA5's 320 KB
// LDS per workgroup makes this possible.  Per 56-chunk (padded to 64):
//   qs = q*exp(Bc), ks = k*exp(-Bc), kT = ks*exp(bend), d = exp(bend)
//   A = qs @ ks^T (tril mask), o = A@v + qs@S, dS = kT^T@v, S = d*S + dS
// All four matmuls are bf16 WMMA.  dir=1 runs the flipped sequence.
// ---------------------------------------------------------------------------
__global__ void __launch_bounds__(256) gla_kernel(const float* __restrict__ qkv,
                                                  const float* __restrict__ gf,
                                                  const float* __restrict__ gb,
                                                  float* __restrict__ of,
                                                  float* __restrict__ ob) {
  extern __shared__ __align__(16) char smem[];
  float*  S    = (float*)smem;                    // [128][256] f32, N-major
  __bf16* qs   = (__bf16*)(smem + 128 * 256 * 4); // [64][256]
  __bf16* ks   = qs + 64 * 256;                   // [64][256]
  __bf16* kTt  = ks + 64 * 256;                   // [256][64]  (A-operand for dS)
  __bf16* vT   = kTt + 256 * 64;                  // [128][64]  (N-major v tile)
  __bf16* Ab   = vT + 128 * 64;                   // [64][64]   (masked A)
  float*  dvec = (float*)(Ab + 64 * 64);          // [256] per-k decay

  int bid = blockIdx.x;
  int dir = bid & 1;
  int vt  = (bid >> 1) & 3;
  int h   = (bid >> 3) & 3;
  int b   = bid >> 5;

  int tid = threadIdx.x, wave = tid >> 5, lane = tid & 31;
  int half = lane >> 4, nl = lane & 15;

  for (int i = tid; i < 128 * 256; i += 256) S[i] = 0.f;
  __syncthreads();

  const float* gsel = dir ? gb : gf;
  float* osel = dir ? ob : of;

  for (int ci = 0; ci < 14; ++ci) {
    // ---- preprocess: this thread owns k-column kk = tid ----
    {
      int kk = tid;
      size_t qcol = (size_t)(h * 256 + kk);
      float acc = 0.f;
      for (int t = 0; t < 56; ++t) {
        int lt = ci * 56 + t;
        int lf = dir ? (783 - lt) : lt;
        size_t row = (size_t)(b * 784 + lf);
        acc += gsel[row * 1024 + h * 256 + kk];
        float ep = __expf(acc), em = __expf(-acc);
        qs[t * 256 + kk] = (__bf16)(qkv[row * 4096 + qcol] * ep);
        ks[t * 256 + kk] = (__bf16)(qkv[row * 4096 + 1024 + qcol] * em);
      }
      float d = __expf(acc);  // exp(bend)
      dvec[kk] = d;
      for (int t = 0; t < 56; ++t)
        kTt[kk * 64 + t] = (__bf16)((float)ks[t * 256 + kk] * d);
      for (int t = 56; t < 64; ++t) {
        qs[t * 256 + kk] = (__bf16)0.f;
        ks[t * 256 + kk] = (__bf16)0.f;
        kTt[kk * 64 + t] = (__bf16)0.f;
      }
    }
    // v tile transposed to N-major: vT[j][t]
    for (int idx = tid; idx < 64 * 128; idx += 256) {
      int t = idx >> 7, j = idx & 127;
      float vv = 0.f;
      if (t < 56) {
        int lt = ci * 56 + t;
        int lf = dir ? (783 - lt) : lt;
        vv = qkv[(size_t)(b * 784 + lf) * 4096 + 2048 + h * 512 + vt * 128 + j];
      }
      vT[j * 64 + t] = (__bf16)vv;
    }
    __syncthreads();

    // ---- A = qs @ ks^T : 64x64, K=256.  2 tiles per wave. ----
#pragma unroll
    for (int ii = 0; ii < 2; ++ii) {
      int ti = wave * 2 + ii;
      int tm = (ti >> 2) * 16, tn = (ti & 3) * 16;
      floatx8 acc = {};
      for (int k0 = 0; k0 < 256; k0 += 32)
        acc = wmma_bf16(ldsFragRM(qs, tm, k0, 256, lane),
                        ldsFragRM(ks, tn, k0, 256, lane), acc);
      int n = tn + nl;
#pragma unroll
      for (int r = 0; r < 8; ++r) {
        int m = tm + r + half * 8;
        float a = (n <= m && m < 56) ? acc[r] : 0.f;  // tril + pad mask
        Ab[m * 64 + n] = (__bf16)a;
      }
    }
    __syncthreads();

    // ---- o = Amask@v + qs@S : 64x128.  4 tiles per wave. ----
#pragma unroll
    for (int ii = 0; ii < 4; ++ii) {
      int ti = wave * 4 + ii;
      int tm = (ti >> 3) * 16, tn = (ti & 7) * 16;
      floatx8 acc = {};
      for (int k0 = 0; k0 < 64; k0 += 32)  // intra-chunk
        acc = wmma_bf16(ldsFragRM(Ab, tm, k0, 64, lane),
                        ldsFragRM(vT, tn, k0, 64, lane), acc);
      for (int k0 = 0; k0 < 256; k0 += 32)  // inter-chunk (reads S)
        acc = wmma_bf16(ldsFragRM(qs, tm, k0, 256, lane),
                        ldsFragF32(S, tn, k0, 256, lane), acc);
      int n = tn + nl;
#pragma unroll
      for (int r = 0; r < 8; ++r) {
        int m = tm + r + half * 8;
        if (m < 56) {
          int lt = ci * 56 + m;
          int lf = dir ? (783 - lt) : lt;
          osel[((size_t)(b * 4 + h) * 784 + lf) * 512 + vt * 128 + n] = acc[r];
        }
      }
    }
    __syncthreads();  // all reads of S complete before update

    // ---- dS = kT^T @ v ; S = d*S + dS : 256x128.  16 tiles per wave. ----
#pragma unroll
    for (int ii = 0; ii < 16; ++ii) {
      int ti = wave * 16 + ii;
      int tm = (ti >> 3) * 16, tn = (ti & 7) * 16;
      floatx8 acc = {};
      for (int k0 = 0; k0 < 64; k0 += 32)
        acc = wmma_bf16(ldsFragRM(kTt, tm, k0, 64, lane),
                        ldsFragRM(vT, tn, k0, 64, lane), acc);
      int n = tn + nl;
#pragma unroll
      for (int r = 0; r < 8; ++r) {
        int kd = tm + r + half * 8;
        float* sp = &S[n * 256 + kd];
        *sp = dvec[kd] * (*sp) + acc[r];
      }
    }
    __syncthreads();
  }
}

// ---------------------------------------------------------------------------
// Combine: o = (rms(of)*gnw + rms(ob)*lnw) * gate, head-layout -> (B,L,DV),
// emitted as bf16 for the final WMMA GEMM.  One block per (b,h,l).
// ---------------------------------------------------------------------------
__global__ void __launch_bounds__(256) combine_kernel(const float* __restrict__ of,
                                                      const float* __restrict__ ob,
                                                      const float* __restrict__ gate,
                                                      const float* __restrict__ gnw,
                                                      const float* __restrict__ lnw,
                                                      __bf16* __restrict__ oc) {
  __shared__ float red[256];
  __shared__ float sfs, sbs;
  int bid = blockIdx.x;  // (b*4+h)*784 + l
  int l = bid % 784, bh = bid / 784;
  int h = bh & 3, b = bh >> 2;
  int tid = threadIdx.x;
  const float* fr = of + (size_t)bid * 512;
  const float* br = ob + (size_t)bid * 512;
  float f0 = fr[tid], f1 = fr[tid + 256];
  float g0 = br[tid], g1 = br[tid + 256];
  red[tid] = f0 * f0 + f1 * f1;
  __syncthreads();
  for (int s = 128; s > 0; s >>= 1) {
    if (tid < s) red[tid] += red[tid + s];
    __syncthreads();
  }
  if (tid == 0) sfs = rsqrtf(red[0] / 512.f + 1e-5f);
  __syncthreads();
  red[tid] = g0 * g0 + g1 * g1;
  __syncthreads();
  for (int s = 128; s > 0; s >>= 1) {
    if (tid < s) red[tid] += red[tid + s];
    __syncthreads();
  }
  if (tid == 0) sbs = rsqrtf(red[0] / 512.f + 1e-5f);
  __syncthreads();
  size_t grow = (size_t)(b * 784 + l) * 2048 + h * 512;
#pragma unroll
  for (int it = 0; it < 2; ++it) {
    int j = tid + it * 256;
    float fv = it ? f1 : f0;
    float bv = it ? g1 : g0;
    float val = (fv * sfs * gnw[j] + bv * sbs * lnw[j]) * gate[grow + j];
    oc[grow + j] = (__bf16)val;
  }
}

// ---------------------------------------------------------------------------
// Host-side launcher
// ---------------------------------------------------------------------------
extern "C" void kernel_launch(void* const* d_in, const int* in_sizes, int n_in,
                              void* d_out, int out_size, void* d_ws, size_t ws_size,
                              hipStream_t stream) {
  (void)in_sizes; (void)n_in; (void)out_size; (void)ws_size;

  const float* x       = (const float*)d_in[0];
  const float* conv_w  = (const float*)d_in[1];
  const float* qkv_w   = (const float*)d_in[2];
  const float* gk_w1   = (const float*)d_in[3];
  const float* gk_w2   = (const float*)d_in[4];
  const float* gk_b2   = (const float*)d_in[5];
  const float* g_w     = (const float*)d_in[6];
  const float* g_b     = (const float*)d_in[7];
  const float* o_w     = (const float*)d_in[8];
  const float* gnorm_w = (const float*)d_in[9];
  const float* lnorm_w = (const float*)d_in[10];
  float* out = (float*)d_out;

  char* ws = (char*)d_ws;
  size_t off = 0;
  auto take = [&](size_t bytes) {
    char* p = ws + off;
    off += (bytes + 255) & ~(size_t)255;
    return p;
  };

  const size_t M = 12544;  // B*L
  float*  xc    = (float*)take(M * 1024 * 4);
  __bf16* xcb   = (__bf16*)take(M * 1024 * 2);
  __bf16* qkvwb = (__bf16*)take((size_t)1024 * 4096 * 2);
  __bf16* gwb   = (__bf16*)take((size_t)1024 * 2048 * 2);
  __bf16* owb   = (__bf16*)take((size_t)2048 * 1024 * 2);
  float*  qkv   = (float*)take(M * 4096 * 4);
  float*  gfb   = (float*)take(M * 1024 * 4);
  float*  gbb   = (float*)take(M * 1024 * 4);
  float*  gate  = (float*)take(M * 2048 * 4);
  float*  ofb   = (float*)take((size_t)16 * 4 * 784 * 512 * 4);
  float*  obb   = (float*)take((size_t)16 * 4 * 784 * 512 * 4);
  __bf16* oc    = (__bf16*)take(M * 2048 * 2);

  // Weight casts to bf16
  cast_bf16_kernel<<<4096, 256, 0, stream>>>(qkv_w, qkvwb);
  cast_bf16_kernel<<<2048, 256, 0, stream>>>(g_w, gwb);
  cast_bf16_kernel<<<2048, 256, 0, stream>>>(o_w, owb);

  // Depthwise conv + SiLU
  conv_silu_kernel<<<16 * 28, 256, 0, stream>>>(x, conv_w, xc, xcb);

  // QKV GEMM: 12544x1024 @ 1024x4096
  gemm_bf16_kernel<<<98 * 64, 256, 0, stream>>>(xcb, qkvwb, nullptr, qkv, 12544, 1024,
                                                4096, 0);
  // Gate logits (low-rank path)
  gk_kernel<<<12544, 256, 0, stream>>>(xc, gk_w1, gk_w2, gk_b2, gfb, gbb);

  // Output gate GEMM + SiLU: 12544x1024 @ 1024x2048
  gemm_bf16_kernel<<<98 * 32, 256, 0, stream>>>(xcb, gwb, g_b, gate, 12544, 1024,
                                                2048, 1);

  // Fused bidirectional chunked GLA scan. 249 KB dynamic LDS (<=320 KB/WGP).
  size_t glaSmem = 128 * 256 * 4 +
                   (size_t)(64 * 256 + 64 * 256 + 256 * 64 + 128 * 64 + 64 * 64) * 2 +
                   256 * 4;
  hipFuncSetAttribute(reinterpret_cast<const void*>(gla_kernel),
                      hipFuncAttributeMaxDynamicSharedMemorySize, (int)glaSmem);
  gla_kernel<<<512, 256, glaSmem, stream>>>(qkv, gfb, gbb, ofb, obb);

  // RMS norms + gating, emit bf16 activations
  combine_kernel<<<16 * 4 * 784, 256, 0, stream>>>(ofb, obb, gate, gnorm_w, lnorm_w, oc);

  // Final projection: 12544x2048 @ 2048x1024 -> fp32 output
  gemm_bf16_kernel<<<98 * 16, 256, 0, stream>>>(oc, owb, nullptr, out, 12544, 2048,
                                                1024, 0);
}
